// RoIPooling_14697378087078
// MI455X (gfx1250) — matrix operand order repaired and checked by reading, not compiled, not run on hardware
//
#include <hip/hip_runtime.h>

typedef __bf16 bf16;
typedef __attribute__((ext_vector_type(16))) __bf16 v16bf;
typedef __attribute__((ext_vector_type(8)))  __bf16 v8bf;
typedef __attribute__((ext_vector_type(8)))  float   v8f;

#define PS   7
#define FMAP 19
#define NB   8
#define NR   128
#define NROI (NB*NR)
#define AROWS 50   // 49 spatial rows + 1 always-zero pad row

// ---------------------------------------------------------------------------
// Fold BN into per-channel scale/bias (conv bias folded in too).
// ---------------------------------------------------------------------------
__global__ void prep_bn(const float* g1,const float* be1,const float* m1,const float* v1,const float* b1,
                        const float* g2,const float* be2,const float* m2,const float* v2,const float* b2,
                        const float* g3,const float* be3,const float* m3,const float* v3,
                        float* s1,float* bi1,float* s2,float* bi2,float* s3,float* bi3){
  int t = threadIdx.x;
  if (t < 256){
    float sc = g1[t]*rsqrtf(v1[t]+1e-5f);
    s1[t]=sc; bi1[t]=b1[t]*sc + be1[t]-m1[t]*sc;
    sc = g2[t]*rsqrtf(v2[t]+1e-5f);
    s2[t]=sc; bi2[t]=b2[t]*sc + be2[t]-m2[t]*sc;
    if (t < 32){ sc = g3[t]*rsqrtf(v3[t]+1e-5f); s3[t]=sc; bi3[t]=be3[t]-m3[t]*sc; }
  }
}

// W[oc][ic][3][3] fp32 -> Wb[tap][oc][ic] bf16 (K-contiguous for WMMA B tiles)
__global__ void prep_w(const float* __restrict__ W, bf16* __restrict__ Wb, int IC){
  int idx = blockIdx.x*256 + threadIdx.x;
  int total = 9*256*IC;
  if (idx >= total) return;
  int tap = idx / (256*IC);
  int rem = idx - tap*256*IC;
  int oc  = rem / IC;
  int ic  = rem - oc*IC;
  Wb[idx] = (bf16)W[(oc*IC + ic)*9 + tap];
}

// ---------------------------------------------------------------------------
// Per-ROI: coords, valid mask, adaptive-pool weights, stable compaction,
// er_c output and keep_count.
// ---------------------------------------------------------------------------
__global__ void roi_prep(const float* __restrict__ roi,
                         float* __restrict__ Wy, float* __restrict__ Wx,
                         int* __restrict__ valid, int* __restrict__ perm,
                         float* __restrict__ er_out, float* __restrict__ keep_out){
  int b = blockIdx.x;          // 0..7
  int r = threadIdx.x;         // 0..127
  const float* rp = roi + (size_t)(b*NR + r)*5;
  int rl[4];
  #pragma unroll
  for (int k=0;k<4;++k) rl[k] = (int)(rp[1+k]*(300.0f/16.0f));   // trunc, coords >= 0
  int c[4];
  #pragma unroll
  for (int k=0;k<4;++k){ int x = rl[k]; x = x<0?0:x; x = x>18?18:x; c[k]=x; }
  int xmin=c[0], ymin=c[1], xmax=c[2], ymax=c[3];
  int v = (xmax>xmin) && (ymax>ymin);
  valid[b*NR+r] = v;

  auto fill = [&](float* dst, int lo, int hi){
    int H = hi - lo + 1;
    for (int i=0;i<PS;++i){
      int s = (i*H)/PS;
      int e = ((i+1)*H + PS-1)/PS;
      int den = e - s; if (den < 1) den = 1;
      float inv = 1.0f/(float)den;
      for (int h=0;h<FMAP;++h){
        float w = (H>=1 && h>=lo+s && h<lo+e) ? inv : 0.0f;
        dst[i*FMAP+h] = w;
      }
    }
  };
  fill(Wy + (size_t)(b*NR+r)*PS*FMAP, ymin, ymax);
  fill(Wx + (size_t)(b*NR+r)*PS*FMAP, xmin, xmax);

  // stable compaction: valid ROIs to front, invalid after (original order kept)
  __shared__ int vf[NR];
  vf[r] = v;
  __syncthreads();
  int rankv=0, ranki=0, total=0;
  for (int j=0;j<NR;++j){
    total += vf[j];
    if (j<r){ rankv += vf[j]; ranki += (vf[j]^1); }
  }
  int dest = v ? rankv : (total + ranki);
  perm[b*NR + dest] = r;

  float* ero = er_out + (size_t)(b*NR + dest)*4;
  #pragma unroll
  for (int k=0;k<4;++k) ero[k] = v ? ((float)rl[k]*(16.0f/300.0f)) : 0.0f;
  if (r==0) keep_out[b] = (float)total;
}

// ---------------------------------------------------------------------------
// Separable ROI pooling: pooled[roi][s=p*7+q][c] (bf16), 16 channels/block.
// Row 49 of each ROI is kept zero (pad row for branch-free conv im2col).
// ---------------------------------------------------------------------------
__global__ void __launch_bounds__(256) pool_kernel(const float* __restrict__ features,
                            const float* __restrict__ Wy, const float* __restrict__ Wx,
                            bf16* __restrict__ pooled){
  int tile = blockIdx.x;     // 0..63 (16 channels each)
  int roi  = blockIdx.y;     // 0..1023
  int b    = roi >> 7;
  int t    = threadIdx.x;
  __shared__ float Fs[16*361];
  __shared__ float Wys[PS*FMAP], Wxs[PS*FMAP];
  __shared__ float Ts[16*PS*FMAP];
  if (t < 133)            Wys[t]      = Wy[(size_t)roi*133 + t];
  else if (t < 266)       Wxs[t-133]  = Wx[(size_t)roi*133 + (t-133)];
  const float* fb = features + ((size_t)b*1024 + tile*16)*361;
  for (int idx=t; idx<16*361; idx+=256) Fs[idx] = fb[idx];
  __syncthreads();
  for (int idx=t; idx<16*PS*FMAP; idx+=256){
    int cl = idx/133; int rem = idx - cl*133; int pp = rem/FMAP; int w = rem - pp*FMAP;
    float s = 0.0f;
    #pragma unroll
    for (int h=0;h<FMAP;++h) s += Wys[pp*FMAP+h]*Fs[cl*361 + h*FMAP + w];
    Ts[idx] = s;
  }
  __syncthreads();
  for (int idx=t; idx<16*49; idx+=256){
    int s9 = idx >> 4; int cl = idx & 15;      // consecutive threads -> consecutive c
    int pp = s9/7, qq = s9 - pp*7;
    float s = 0.0f;
    #pragma unroll
    for (int w=0;w<FMAP;++w) s += Ts[cl*133 + pp*FMAP + w]*Wxs[qq*FMAP+w];
    pooled[((size_t)roi*AROWS + s9)*1024 + tile*16 + cl] = (bf16)s;
  }
  if (t < 16) pooled[((size_t)roi*AROWS + 49)*1024 + tile*16 + t] = (bf16)0.0f;
}

// ---------------------------------------------------------------------------
// 3x3 pad-1 conv over 7x7 as implicit GEMM with bf16 WMMA.
// Block = (roi, 64-oc tile); M=64 (49 real). K staged 64-wide, LDS
// double-buffered -> 1 barrier per 4 WMMAs, branch-free (zero pad row).
// Asrc: [roi][50][IC] bf16.  Wb: [tap][oc][IC] bf16.  Out: [roi][50][256] bf16.
// ---------------------------------------------------------------------------
__global__ void __launch_bounds__(256) conv3x3_wmma(const bf16* __restrict__ Asrc, int IC, int logc,
                         const bf16* __restrict__ Wb,
                         const float* __restrict__ scale, const float* __restrict__ bias,
                         bf16* __restrict__ Out){
  int roi = blockIdx.x >> 2;
  int ocb = (blockIdx.x & 3) * 64;
  int t    = threadIdx.x;
  int lane = t & 31;
  int wave = t >> 5;
  int mi   = wave & 3;            // M sub-tile 0..3
  int ni0  = (wave >> 2) * 2;     // two N sub-tiles per wave
  __shared__ __align__(32) bf16 Alds[2][64*64];
  __shared__ __align__(32) bf16 Blds[2][64*64];   // n-major: [n][k]
  v8f acc0 = {}; v8f acc1 = {};

  int sm  = t >> 2;               // staged row 0..63 (spatial for A, oc for B)
  int kch = (t & 3) * 8;          // first 8-element chunk within 64-wide K
  int p = sm/7, q = sm - p*7;
  int chunks = IC >> 6;           // 64-wide K chunks per tap
  int cmask  = chunks - 1;
  int nsteps = 9 << logc;         // 9 taps * chunks

  const bf16* arowbase = Asrc + (size_t)roi*AROWS*IC;

  auto stage = [&](int it, int buf){
    int tap = it >> logc;
    int icb = it & cmask;
    int dy = tap/3 - 1, dx = tap - (tap/3)*3 - 1;
    int ps = p + dy, qs = q + dx;
    bool arow = (sm < 49) && (ps>=0) && (ps<7) && (qs>=0) && (qs<7);
    int srow = arow ? (ps*7+qs) : 49;                 // row 49 is all zeros
    const bf16* ab = arowbase + (size_t)srow*IC + icb*64;
    const bf16* bb = Wb + ((size_t)tap*256 + ocb + sm)*IC + icb*64;
    *(v8bf*)&Alds[buf][sm*64 + kch]      = *(const v8bf*)(ab + kch);
    *(v8bf*)&Alds[buf][sm*64 + kch + 32] = *(const v8bf*)(ab + kch + 32);
    *(v8bf*)&Blds[buf][sm*64 + kch]      = *(const v8bf*)(bb + kch);
    *(v8bf*)&Blds[buf][sm*64 + kch + 32] = *(const v8bf*)(bb + kch + 32);
  };

  int ar  = 16*mi + (lane & 15);
  int kb  = (lane >> 4) * 16;     // A half-select (contiguous k, matches B)
  int n0  = 16*ni0 + (lane & 15);
  int kbb = (lane >> 4) * 16;     // B half-select

  auto compute = [&](int buf){
    #pragma unroll
    for (int kk=0; kk<2; ++kk){
      int ka = kk*32;
      v8bf alo = *(const v8bf*)&Alds[buf][ar*64 + ka + kb];
      v8bf ahi = *(const v8bf*)&Alds[buf][ar*64 + ka + kb + 8];
      v16bf afrag;
      #pragma unroll
      for (int j=0;j<8;++j){ afrag[j]=alo[j]; afrag[8+j]=ahi[j]; }
      v16bf bfrag0 = *(const v16bf*)&Blds[buf][ n0     *64 + ka + kbb];
      v16bf bfrag1 = *(const v16bf*)&Blds[buf][(n0+16)*64 + ka + kbb];
      acc0 = __builtin_amdgcn_wmma_f32_16x16x32_bf16(false, afrag, false, bfrag0,
                                                     (short)0, acc0, false, false);
      acc1 = __builtin_amdgcn_wmma_f32_16x16x32_bf16(false, afrag, false, bfrag1,
                                                     (short)0, acc1, false, false);
    }
  };

  stage(0, 0);
  __syncthreads();
  for (int it=0; it<nsteps; ++it){
    int cur = it & 1;
    if (it+2 < nsteps){   // prefetch weight tile two steps ahead into GL2
      int tp = (it+2) >> logc, ib = (it+2) & cmask;
      __builtin_prefetch(Wb + ((size_t)tp*256 + ocb + sm)*IC + ib*64, 0, 1);
    }
    if (it+1 < nsteps) stage(it+1, cur^1);
    compute(cur);
    __syncthreads();
  }

  // Epilogue: fused BN scale/bias + ReLU, bf16 store; keep pad row 49 zero.
  int nl = 16*ni0 + (lane & 15);
  #pragma unroll
  for (int vv=0; vv<8; ++vv){
    int m = 16*mi + vv + 8*(lane >> 4);
    if (m < 49){
      size_t ob = ((size_t)roi*AROWS + m)*256 + ocb;
      float y0 = acc0[vv]*scale[ocb+nl]      + bias[ocb+nl];
      float y1 = acc1[vv]*scale[ocb+nl+16]   + bias[ocb+nl+16];
      Out[ob + nl]      = (bf16)(y0 > 0.0f ? y0 : 0.0f);
      Out[ob + nl + 16] = (bf16)(y1 > 0.0f ? y1 : 0.0f);
    }
  }
  if (t < 64) Out[((size_t)roi*AROWS + 49)*256 + ocb + t] = (bf16)0.0f;
}

// ---------------------------------------------------------------------------
// 1x1 conv 256->32 + BN + ReLU, written directly to the compacted slot.
// Block = (batch, dest slot). Invalid slots -> zeros.
// ---------------------------------------------------------------------------
__global__ void __launch_bounds__(256) conv3_out(const bf16* __restrict__ act2, const float* __restrict__ W3,
                          const float* __restrict__ s3, const float* __restrict__ bi3,
                          const int* __restrict__ valid, const int* __restrict__ perm,
                          float* __restrict__ out){
  int blk = blockIdx.x;
  int b = blk >> 7;
  int d = blk & 127;
  int r = perm[b*NR + d];
  int v = valid[b*NR + r];
  float* ob = out + 4096 + (size_t)(b*NR + d)*32*49;
  int t = threadIdx.x;
  if (!v){
    for (int idx=t; idx<32*49; idx+=256) ob[idx] = 0.0f;
    return;
  }
  int roi = b*NR + r;
  const bf16* ab = act2 + (size_t)roi*AROWS*256;
  for (int idx=t; idx<32*49; idx+=256){
    int oc = idx/49, s = idx - oc*49;
    const bf16*  row = ab + s*256;
    const float* w   = W3 + oc*256;
    float acc = 0.0f;
    for (int ic=0; ic<256; ++ic) acc += (float)row[ic]*w[ic];
    float y = acc*s3[oc] + bi3[oc];
    ob[idx] = y > 0.0f ? y : 0.0f;
  }
}

// ---------------------------------------------------------------------------
extern "C" void kernel_launch(void* const* d_in, const int* in_sizes, int n_in,
                              void* d_out, int out_size, void* d_ws, size_t ws_size,
                              hipStream_t stream){
  const float* roi  = (const float*)d_in[0];
  const float* feat = (const float*)d_in[1];
  const float* W1   = (const float*)d_in[2];
  const float* b1   = (const float*)d_in[3];
  const float* g1   = (const float*)d_in[4];
  const float* be1  = (const float*)d_in[5];
  const float* m1   = (const float*)d_in[6];
  const float* v1   = (const float*)d_in[7];
  const float* W2   = (const float*)d_in[8];
  const float* b2   = (const float*)d_in[9];
  const float* g2   = (const float*)d_in[10];
  const float* be2  = (const float*)d_in[11];
  const float* m2   = (const float*)d_in[12];
  const float* v2   = (const float*)d_in[13];
  const float* W3   = (const float*)d_in[14];
  const float* g3   = (const float*)d_in[15];
  const float* be3  = (const float*)d_in[16];
  const float* m3   = (const float*)d_in[17];
  const float* v3   = (const float*)d_in[18];

  char* ws = (char*)d_ws;
  float* s1   = (float*)(ws + 0);
  float* bi1  = (float*)(ws + 1024);
  float* s2   = (float*)(ws + 2048);
  float* bi2  = (float*)(ws + 3072);
  float* s3   = (float*)(ws + 4096);
  float* bi3  = (float*)(ws + 4352);
  int*   dval = (int*)  (ws + 4608);
  int*   dperm= (int*)  (ws + 8704);
  float* dWy  = (float*)(ws + 12800);
  float* dWx  = (float*)(ws + 557568);
  bf16*  W1b  = (bf16*) (ws + 1102336);
  bf16*  W2b  = (bf16*) (ws + 5820928);
  bf16*  pooled = (bf16*)(ws + 7000576);                        // 1024*50*1024 bf16
  bf16*  act1   = (bf16*)(ws + 7000576 + (size_t)104857600);    // 1024*50*256 bf16
  bf16*  act2   = (bf16*)(ws + 7000576 + (size_t)104857600 + (size_t)26214400);

  float* out     = (float*)d_out;
  float* keepOut = out + 4096 + (size_t)NROI*32*49;

  prep_bn<<<1,256,0,stream>>>(g1,be1,m1,v1,b1, g2,be2,m2,v2,b2, g3,be3,m3,v3,
                              s1,bi1,s2,bi2,s3,bi3);
  prep_w<<<(9*256*1024)/256,256,0,stream>>>(W1, W1b, 1024);
  prep_w<<<(9*256*256)/256, 256,0,stream>>>(W2, W2b, 256);
  roi_prep<<<NB,NR,0,stream>>>(roi, dWy, dWx, dval, dperm, out, keepOut);
  pool_kernel<<<dim3(64, NROI),256,0,stream>>>(feat, dWy, dWx, pooled);
  conv3x3_wmma<<<NROI*4,256,0,stream>>>(pooled, 1024, 4, W1b, s1, bi1, act1);
  conv3x3_wmma<<<NROI*4,256,0,stream>>>(act1,    256, 2, W2b, s2, bi2, act2);
  conv3_out<<<NROI,256,0,stream>>>(act2, W3, s3, bi3, dval, dperm, out);
}
